// CrossAttention_37950331027766
// MI455X (gfx1250) — matrix operand order repaired and checked by reading, not compiled, not run on hardware
//
#include <hip/hip_runtime.h>

// ---------------------------------------------------------------------------
// Problem constants (reference: DIM=256, HEADS=8, DHEAD=64, b=1, n=6, Q=900, K=1408)
// ---------------------------------------------------------------------------
#define DIMD   256
#define HEADS  8
#define DHEAD  64
#define INNERD 512
#define NCH    6
#define QN     900
#define KN     1408

typedef __attribute__((ext_vector_type(16))) __bf16 bf16x16;
typedef __attribute__((ext_vector_type(8)))  float  f32x8;

union FragAB { bf16x16 v; unsigned u[8]; };

static __device__ __forceinline__ unsigned short f2bf_bits(float f) {
    unsigned u = __float_as_uint(f);
    unsigned r = u + 0x7FFFu + ((u >> 16) & 1u);   // round-to-nearest-even
    return (unsigned short)(r >> 16);
}

static __device__ __forceinline__ f32x8 zero8() {
    f32x8 z;
#pragma unroll
    for (int i = 0; i < 8; ++i) z[i] = 0.0f;
    return z;
}

static __device__ __forceinline__ f32x8 wmma_bf16(const FragAB& a, const FragAB& b, f32x8 c) {
    return __builtin_amdgcn_wmma_f32_16x16x32_bf16(false, a.v, false, b.v, (short)0, c, false, false);
}

// Async global->LDS copy of one 32-row x 128-byte tile (bf16 payload), 128 threads,
// 2 x 16B issues per thread. LDS rows padded to 144 B for bank spread; 16B aligned.
// Tracked by ASYNCcnt (CDNA5 async path).
static __device__ __forceinline__ void async_tile32(const unsigned short* g0,
                                                    unsigned short (*ldsrows)[72],
                                                    int tid) {
#pragma unroll
    for (int i = 0; i < 2; ++i) {
        int off = i * 2048 + tid * 16;
        int row = off >> 7;
        int col = off & 127;
        const char* g = (const char*)g0 + (size_t)row * (INNERD * 2) + col;
        unsigned loff = (unsigned)(uintptr_t)((const char*)&ldsrows[row][0] + col);
        asm volatile("global_load_async_to_lds_b128 %0, %1, off"
                     :: "v"(loff), "v"(g)
                     : "memory");
    }
}

// ---------------------------------------------------------------------------
// Generic tiled GEMM:  out = act(A(f32)[MxK] @ W(f32)[KxN] + bias) (+resid)
// 64x64 tile per 128-thread block (4 waves, one 16-row strip each), BK=32.
// A and W staged to LDS as bf16 (W transposed so B-fragment pairs are contiguous).
// ---------------------------------------------------------------------------
template <bool OUTBF16, bool GELU>
__global__ __launch_bounds__(128) void gemm_ws(const float* __restrict__ A,
                                               const float* __restrict__ W,
                                               const float* __restrict__ bias,
                                               const float* __restrict__ resid,
                                               void* __restrict__ out,
                                               int M, int N, int Kd) {
    __shared__ unsigned short As[64][36];   // [row][k], +4 halfword pad
    __shared__ unsigned short Bs[64][36];   // [col][k] (transposed), +4 pad

    const int bm = blockIdx.x * 64, bn = blockIdx.y * 64;
    const int tid = threadIdx.x;
    const int wave = tid >> 5, lane = tid & 31;
    const int kg = lane >> 4, ln = lane & 15;

    f32x8 acc[4];
#pragma unroll
    for (int c = 0; c < 4; ++c) acc[c] = zero8();

    for (int k0 = 0; k0 < Kd; k0 += 32) {
        // ---- stage A tile (64x32 f32 -> bf16) ----
#pragma unroll
        for (int i = 0; i < 16; ++i) {
            int idx = i * 128 + tid;
            int r = idx >> 5, c = idx & 31;
            int gr = bm + r; if (gr >= M) gr = M - 1;
            As[r][c] = f2bf_bits(A[(size_t)gr * Kd + k0 + c]);
        }
        // ---- stage W tile (32x64 f32 -> bf16, transposed into [col][k]) ----
#pragma unroll
        for (int i = 0; i < 16; ++i) {
            int idx = i * 128 + tid;
            int kk = idx >> 6, c = idx & 63;
            Bs[c][kk] = f2bf_bits(W[(size_t)(k0 + kk) * N + bn + c]);
        }
        if (k0 + 32 < Kd) {  // prefetch next tiles (global_prefetch_b8)
            __builtin_prefetch(&A[(size_t)(bm + (tid >> 1)) * Kd + k0 + 32], 0, 0);
            __builtin_prefetch(&W[(size_t)(k0 + 32 + (tid & 31)) * N + bn], 0, 0);
        }
        __syncthreads();

        // A fragment: 16x32 bf16, lane = (kg, m=ln); VGPR v holds K pair
        FragAB af;
#pragma unroll
        for (int v = 0; v < 8; ++v) {
            int kp = ((v < 4) ? 2 * v : 2 * v + 8) + 8 * kg;
            af.u[v] = *(const unsigned*)&As[wave * 16 + ln][kp];
        }
#pragma unroll
        for (int c = 0; c < 4; ++c) {
            // B fragment: 32x16 bf16, lane = (kg, n=ln); VGPR v holds K pair (v<4?2v:2v+8)+8kg
            FragAB bf;
#pragma unroll
            for (int v = 0; v < 8; ++v) {
                int kp = ((v < 4) ? 2 * v : 2 * v + 8) + 8 * kg;
                bf.u[v] = *(const unsigned*)&Bs[c * 16 + ln][kp];
            }
            acc[c] = wmma_bf16(af, bf, acc[c]);
        }
        __syncthreads();
    }

    // ---- epilogue: bias / gelu / residual; C layout: row = r + 8*kg, col = ln ----
#pragma unroll
    for (int c = 0; c < 4; ++c) {
        int col = bn + c * 16 + ln;
        float bs = bias ? bias[col] : 0.0f;
#pragma unroll
        for (int r = 0; r < 8; ++r) {
            int row = bm + wave * 16 + r + 8 * kg;
            if (row < M) {
                float val = acc[c][r] + bs;
                if (GELU) val = 0.5f * val * (1.0f + erff(val * 0.70710678118654752f));
                if (resid) val += resid[(size_t)row * N + col];
                if (OUTBF16) ((unsigned short*)out)[(size_t)row * N + col] = f2bf_bits(val);
                else         ((float*)out)[(size_t)row * N + col] = val;
            }
        }
    }
}

// ---------------------------------------------------------------------------
// Flash attention (partial over one n-chunk).
// grid = (ceil(Q/64), HEADS, NCH); block = 128 (4 waves x 16 q-rows).
// KG/KA/V 32-key tiles are staged once per block into double-buffered LDS via
// GLOBAL_LOAD_ASYNC_TO_LDS_B128 (all four waves consume the same tiles),
// prefetching the next tile set while computing on the current one.
// Scores: S = sg * (Qg Kg^T) + sa * (Qa Ka^T), masked; online softmax; O += P V.
// Partials (m, l, O-unnormalized) written per chunk; combined later.
// ---------------------------------------------------------------------------
__global__ __launch_bounds__(128) void attn_partial(const unsigned short* __restrict__ QG,
                                                    const unsigned short* __restrict__ QAb,
                                                    const unsigned short* __restrict__ KG,
                                                    const unsigned short* __restrict__ KA,
                                                    const unsigned short* __restrict__ VVp,
                                                    const unsigned char* __restrict__ mask,
                                                    const float* __restrict__ scale_g,
                                                    const float* __restrict__ scale_a,
                                                    float* __restrict__ Opart,
                                                    float* __restrict__ Mpart,
                                                    float* __restrict__ Lpart) {
    const int head = blockIdx.y, nn = blockIdx.z;
    const int tid = threadIdx.x;
    const int wave = tid >> 5, lane = tid & 31;
    const int kg = lane >> 4, ln = lane & 15;
    const int qbase = blockIdx.x * 64 + wave * 16;
    const float sg = scale_g[head], sa = scale_a[head];
    const int h64 = head * DHEAD;

    __shared__ __align__(16) unsigned short KGt[2][32][72];  // [buf][key][d], 144B rows
    __shared__ __align__(16) unsigned short KAt[2][32][72];
    __shared__ __align__(16) unsigned short Vt[2][32][72];
    __shared__ unsigned short Pld[4][16][36];                // per-wave P tile

    const unsigned short* KGbase = KG + ((size_t)nn * KN) * INNERD + h64;
    const unsigned short* KAbase = KA + ((size_t)nn * KN) * INNERD + h64;
    const unsigned short* Vbase  = VVp + ((size_t)nn * KN) * INNERD + h64;

    int qrow = qbase + ln; if (qrow > QN - 1) qrow = QN - 1;

    // Q fragments (A-matrix 16x32 layout), dc = d-chunk of 32 within DHEAD=64
    FragAB qgf[2], qaf[2];
#pragma unroll
    for (int dc = 0; dc < 2; ++dc) {
#pragma unroll
        for (int v = 0; v < 8; ++v) {
            int kp = ((v < 4) ? 2 * v : 2 * v + 8) + 8 * kg + dc * 32;
            qaf[dc].u[v] = *(const unsigned*)&QAb[(size_t)qrow * INNERD + h64 + kp];
            qgf[dc].u[v] = *(const unsigned*)&QG[((size_t)nn * QN + qrow) * INNERD + h64 + kp];
        }
    }

    // prologue: stage the first 32-key tile set into buffer 0
    async_tile32(KGbase, KGt[0], tid);
    async_tile32(KAbase, KAt[0], tid);
    async_tile32(Vbase,  Vt[0],  tid);

    f32x8 o[4];
    float mx[8], lsum[8];
#pragma unroll
    for (int c = 0; c < 4; ++c) o[c] = zero8();
#pragma unroll
    for (int r = 0; r < 8; ++r) { mx[r] = -1e30f; lsum[r] = 0.0f; }

    for (int kb = 0; kb < KN; kb += 32) {
        const int p = (kb >> 5) & 1;
        __syncthreads();                       // prior readers of buf p^1 are done
        if (kb + 32 < KN) {                    // prefetch next tile set into buf p^1
            const size_t nb = (size_t)(kb + 32) * INNERD;
            async_tile32(KGbase + nb, KGt[p ^ 1], tid);
            async_tile32(KAbase + nb, KAt[p ^ 1], tid);
            async_tile32(Vbase + nb,  Vt[p ^ 1],  tid);
            asm volatile("s_wait_asynccnt 0x6" ::: "memory");  // current set landed
        } else {
            asm volatile("s_wait_asynccnt 0x0" ::: "memory");
        }
        __syncthreads();                       // LDS tiles visible block-wide

        float Sv[2][8];
#pragma unroll
        for (int s = 0; s < 2; ++s) {
            const int keyrow = s * 16 + ln;    // B column = key (within tile)
            FragAB kgf[2], kaf[2];
#pragma unroll
            for (int dc = 0; dc < 2; ++dc)
#pragma unroll
                for (int v = 0; v < 8; ++v) {
                    int kp = dc * 32 + ((v < 4) ? 2 * v : 2 * v + 8) + 8 * kg;   // K dim = dhead index
                    kgf[dc].u[v] = *(const unsigned*)&KGt[p][keyrow][kp];
                    kaf[dc].u[v] = *(const unsigned*)&KAt[p][keyrow][kp];
                }
            f32x8 zg = zero8(), za = zero8();
            zg = wmma_bf16(qgf[0], kgf[0], zg);
            zg = wmma_bf16(qgf[1], kgf[1], zg);
            za = wmma_bf16(qaf[0], kaf[0], za);
            za = wmma_bf16(qaf[1], kaf[1], za);
            const bool mk = mask[(size_t)nn * KN + kb + s * 16 + ln] != 0;
#pragma unroll
            for (int r = 0; r < 8; ++r) {
                float t = sg * zg[r] + sa * za[r];
                Sv[s][r] = mk ? -1e30f : t;
            }
        }

        // online softmax: row stats live in C-layout register space (row = r + 8*kg)
        float pr[2][8];
#pragma unroll
        for (int r = 0; r < 8; ++r) {
            float t = fmaxf(Sv[0][r], Sv[1][r]);
#pragma unroll
            for (int m = 1; m < 16; m <<= 1) t = fmaxf(t, __shfl_xor(t, m, 32));
            float nm = fmaxf(mx[r], t);
            float sc = __expf(mx[r] - nm);
            mx[r] = nm;
            lsum[r] *= sc;
#pragma unroll
            for (int c = 0; c < 4; ++c) o[c][r] *= sc;
        }
#pragma unroll
        for (int r = 0; r < 8; ++r) {
            pr[0][r] = __expf(Sv[0][r] - mx[r]);
            pr[1][r] = __expf(Sv[1][r] - mx[r]);
            float t = pr[0][r] + pr[1][r];
#pragma unroll
            for (int m = 1; m < 16; m <<= 1) t += __shfl_xor(t, m, 32);
            lsum[r] += t;
        }

        // reshuffle P from C-layout to A-fragment layout via per-wave LDS tile
#pragma unroll
        for (int s = 0; s < 2; ++s)
#pragma unroll
            for (int r = 0; r < 8; ++r)
                Pld[wave][r + 8 * kg][s * 16 + ln] = f2bf_bits(pr[s][r]);
        asm volatile("s_wait_dscnt 0x0" ::: "memory");

        FragAB pf;
#pragma unroll
        for (int v = 0; v < 8; ++v) {
            int kp = ((v < 4) ? 2 * v : 2 * v + 8) + 8 * kg;
            pf.u[v] = *(const unsigned*)&Pld[wave][ln][kp];
        }

        // O += P @ V ; B = V[key][d] from the staged LDS tile
#pragma unroll
        for (int c = 0; c < 4; ++c) {
            FragAB vf;
#pragma unroll
            for (int v = 0; v < 8; ++v) {
                int kr = ((v < 4) ? 2 * v : 2 * v + 8) + 8 * kg;
                unsigned lo = Vt[p][kr][c * 16 + ln];
                unsigned hi = Vt[p][kr + 1][c * 16 + ln];
                vf.u[v] = lo | (hi << 16);
            }
            o[c] = wmma_bf16(pf, vf, o[c]);
        }
    }

    // store partials (unnormalized O + per-row m, l)
    const int slab = nn * HEADS + head;
#pragma unroll
    for (int r = 0; r < 8; ++r) {
        int row = qbase + r + 8 * kg;
        if (row < QN) {
            if (ln == 0) {
                Mpart[(size_t)slab * QN + row] = mx[r];
                Lpart[(size_t)slab * QN + row] = lsum[r];
            }
#pragma unroll
            for (int c = 0; c < 4; ++c)
                Opart[((size_t)slab * QN + row) * DHEAD + c * 16 + ln] = o[c][r];
        }
    }
}

// ---------------------------------------------------------------------------
// Combine the NCH per-chunk flash partials into the final attention output
// laid out as Aout[q][head*64 + d]  (== reference transpose to (Q, h*dh)).
// ---------------------------------------------------------------------------
__global__ __launch_bounds__(256) void attn_combine(const float* __restrict__ Opart,
                                                    const float* __restrict__ Mpart,
                                                    const float* __restrict__ Lpart,
                                                    float* __restrict__ Aout) {
    int idx = blockIdx.x * 256 + threadIdx.x;
    if (idx >= QN * INNERD) return;
    int q = idx >> 9;
    int rem = idx & 511;
    int head = rem >> 6;
    int dd = rem & 63;

    float mstar = -1e30f;
#pragma unroll
    for (int i = 0; i < NCH; ++i)
        mstar = fmaxf(mstar, Mpart[(size_t)(i * HEADS + head) * QN + q]);
    float lsum = 0.0f, osum = 0.0f;
#pragma unroll
    for (int i = 0; i < NCH; ++i) {
        size_t s = (size_t)(i * HEADS + head) * QN + q;
        float w = __expf(Mpart[s] - mstar);
        lsum += Lpart[s] * w;
        osum += Opart[s * DHEAD + dd] * w;
    }
    Aout[(size_t)q * INNERD + head * DHEAD + dd] = osum / fmaxf(lsum, 1e-20f);
}

// ---------------------------------------------------------------------------
// Row LayerNorm over N=256 (one row per 256-thread block).
// ---------------------------------------------------------------------------
__global__ __launch_bounds__(256) void ln_kernel(const float* __restrict__ x,
                                                 const float* __restrict__ g,
                                                 const float* __restrict__ b,
                                                 float* __restrict__ y,
                                                 int N) {
    __shared__ float r1[8], r2[8];
    const int row = blockIdx.x;
    const int tid = threadIdx.x;
    float v = x[(size_t)row * N + tid];
    float s1 = v, s2 = v * v;
#pragma unroll
    for (int m = 16; m > 0; m >>= 1) {
        s1 += __shfl_xor(s1, m, 32);
        s2 += __shfl_xor(s2, m, 32);
    }
    if ((tid & 31) == 0) { r1[tid >> 5] = s1; r2[tid >> 5] = s2; }
    __syncthreads();
    if (tid == 0) {
        float a = 0.0f, c = 0.0f;
#pragma unroll
        for (int i = 0; i < 8; ++i) { a += r1[i]; c += r2[i]; }
        r1[0] = a; r2[0] = c;
    }
    __syncthreads();
    float mean = r1[0] / (float)N;
    float var = r2[0] / (float)N - mean * mean;
    y[(size_t)row * N + tid] = (v - mean) * rsqrtf(var + 1e-5f) * g[tid] + b[tid];
}

// ---------------------------------------------------------------------------
// Host-side orchestration
// ---------------------------------------------------------------------------
extern "C" void kernel_launch(void* const* d_in, const int* in_sizes, int n_in,
                              void* d_out, int out_size, void* d_ws, size_t ws_size,
                              hipStream_t stream) {
    (void)in_sizes; (void)n_in; (void)out_size; (void)ws_size;

    const float* k_g  = (const float*)d_in[0];
    const float* q_g  = (const float*)d_in[1];
    const float* k_a  = (const float*)d_in[2];
    const float* q_a  = (const float*)d_in[3];
    const float* vin  = (const float*)d_in[4];
    const unsigned char* mask = (const unsigned char*)d_in[5];
    const float* W_qg = (const float*)d_in[6];  const float* b_qg = (const float*)d_in[7];
    const float* W_kg = (const float*)d_in[8];  const float* b_kg = (const float*)d_in[9];
    const float* W_qa = (const float*)d_in[10]; const float* b_qa = (const float*)d_in[11];
    const float* W_ka = (const float*)d_in[12]; const float* b_ka = (const float*)d_in[13];
    const float* W_v  = (const float*)d_in[14]; const float* b_v  = (const float*)d_in[15];
    const float* W_o  = (const float*)d_in[16]; const float* b_o  = (const float*)d_in[17];
    const float* ln1_g = (const float*)d_in[18]; const float* ln1_b = (const float*)d_in[19];
    const float* W_m1 = (const float*)d_in[20]; const float* b_m1 = (const float*)d_in[21];
    const float* W_m2 = (const float*)d_in[22]; const float* b_m2 = (const float*)d_in[23];
    const float* ln2_g = (const float*)d_in[24]; const float* ln2_b = (const float*)d_in[25];
    const float* scale_g = (const float*)d_in[26];
    const float* scale_a = (const float*)d_in[27];

    // workspace carve-out
    char* p = (char*)d_ws;
    auto carve = [&](size_t bytes) -> void* {
        void* r = (void*)p;
        p += (bytes + 255) & ~(size_t)255;
        return r;
    };
    unsigned short* QG  = (unsigned short*)carve((size_t)NCH * QN * INNERD * 2);
    unsigned short* QAb = (unsigned short*)carve((size_t)QN * INNERD * 2);
    unsigned short* KG  = (unsigned short*)carve((size_t)NCH * KN * INNERD * 2);
    unsigned short* KA  = (unsigned short*)carve((size_t)NCH * KN * INNERD * 2);
    unsigned short* VV  = (unsigned short*)carve((size_t)NCH * KN * INNERD * 2);
    float* Opart = (float*)carve((size_t)NCH * HEADS * QN * DHEAD * 4);
    float* Mpart = (float*)carve((size_t)NCH * HEADS * QN * 4);
    float* Lpart = (float*)carve((size_t)NCH * HEADS * QN * 4);
    float* Aout  = (float*)carve((size_t)QN * INNERD * 4);
    float* Zpre  = (float*)carve((size_t)QN * DIMD * 4);
    float* Z     = (float*)carve((size_t)QN * DIMD * 4);
    float* Hbuf  = (float*)carve((size_t)QN * INNERD * 4);
    float* Z2    = (float*)carve((size_t)QN * DIMD * 4);

    const dim3 blk(128);
    const int MQG = NCH * QN;   // 5400
    const int MKK = NCH * KN;   // 8448

    // ---- projections (f32 -> bf16 activations via WMMA) ----
    gemm_ws<true, false><<<dim3((MQG + 63) / 64, INNERD / 64), blk, 0, stream>>>(
        q_g, W_qg, b_qg, nullptr, QG, MQG, INNERD, DIMD);
    gemm_ws<true, false><<<dim3((QN + 63) / 64, INNERD / 64), blk, 0, stream>>>(
        q_a, W_qa, b_qa, nullptr, QAb, QN, INNERD, DIMD);
    gemm_ws<true, false><<<dim3((MKK + 63) / 64, INNERD / 64), blk, 0, stream>>>(
        k_g, W_kg, b_kg, nullptr, KG, MKK, INNERD, DIMD);
    gemm_ws<true, false><<<dim3((MKK + 63) / 64, INNERD / 64), blk, 0, stream>>>(
        k_a, W_ka, b_ka, nullptr, KA, MKK, INNERD, DIMD);
    gemm_ws<true, false><<<dim3((MKK + 63) / 64, INNERD / 64), blk, 0, stream>>>(
        vin, W_v, b_v, nullptr, VV, MKK, INNERD, DIMD);

    // ---- flash attention: per-chunk partials (async-LDS staged), then combine ----
    attn_partial<<<dim3((QN + 63) / 64, HEADS, NCH), blk, 0, stream>>>(
        QG, QAb, KG, KA, VV, mask, scale_g, scale_a, Opart, Mpart, Lpart);
    attn_combine<<<dim3((QN * INNERD + 255) / 256), 256, 0, stream>>>(
        Opart, Mpart, Lpart, Aout);

    // ---- output projection + skip, LN1, MLP (exact GELU), residual, LN2 ----
    gemm_ws<false, false><<<dim3((QN + 63) / 64, DIMD / 64), blk, 0, stream>>>(
        Aout, W_o, b_o, q_a /* skip */, Zpre, QN, DIMD, INNERD);
    ln_kernel<<<QN, 256, 0, stream>>>(Zpre, ln1_g, ln1_b, Z, DIMD);
    gemm_ws<false, true><<<dim3((QN + 63) / 64, INNERD / 64), blk, 0, stream>>>(
        Z, W_m1, b_m1, nullptr, Hbuf, QN, INNERD, DIMD);
    gemm_ws<false, false><<<dim3((QN + 63) / 64, DIMD / 64), blk, 0, stream>>>(
        Hbuf, W_m2, b_m2, Z /* residual */, Z2, QN, DIMD, INNERD);
    ln_kernel<<<QN, 256, 0, stream>>>(Z2, ln2_g, ln2_b, (float*)d_out, DIMD);
}
